// NaiveSelfAttSITAggregator_66391604461715
// MI455X (gfx1250) — matrix operand order repaired and checked by reading, not compiled, not run on hardware
//
#include <hip/hip_runtime.h>

// ---------------------------------------------------------------------------
// Causal self-attention + post-softmax pad mask + concat(self) for MI455X.
// B=8, T=2048, D=768.  Compute-bound (~103 GFLOP vs 150MB traffic), so both
// GEMMs run on v_wmma_f32_16x16x32_bf16 (bf16 inputs, f32 accumulate).
//
// Block: 256 threads = 8 waves = 2 row-groups(16 queries) x 4 j-slices.
// Flash-style online softmax over 64-key chunks staged in LDS (bf16, both
// row-major for QK^T and feature-major for P*V).  ~260KB LDS / WGP (<=320KB).
// ---------------------------------------------------------------------------

typedef __attribute__((ext_vector_type(16))) __bf16 v16bf;
typedef __attribute__((ext_vector_type(8)))  __bf16 v8bf;
typedef __attribute__((ext_vector_type(4)))  __bf16 v4bf;
typedef __attribute__((ext_vector_type(8)))  float  v8f;

union Frag { v16bf v; v8bf h[2]; };

#define T_SEQ   2048
#define DH      768
#define KROW    1552                 // 768 bf16 + 8 pad  (16B aligned, bank-skewed)
#define VTROW   144                  // 64 bf16 + 8 pad
#define PROW    144

// LDS layout (bytes)
#define OFF_Q     0                  // 32  * KROW  = 49664
#define OFF_K     49664              // 64  * KROW  = 99328
#define OFF_VT    148992             // 768 * VTROW = 110592
#define OFF_P     259584             // 2 * 16 * PROW = 4608
#define OFF_KM    264192             // 64 f32
#define OFF_QM    264448             // 32 f32
#define OFF_SMAX  264576             // 2*4*16 f32
#define OFF_SSUM  265088             // 2*4*16 f32
#define OFF_SCB   265600             // 32 f32
#define OFF_MRUN  265728             // 32 f32
#define OFF_LRUN  265856             // 32 f32
#define SMEM_BYTES 265984

__device__ __forceinline__ v8f wmma_bf16(const Frag& a, const Frag& b, v8f c) {
  return __builtin_amdgcn_wmma_f32_16x16x32_bf16(false, a.v, false, b.v,
                                                 (short)0, c, false, false);
}

__global__ __launch_bounds__(256, 1)
void attn_kernel(const float* __restrict__ text, const float* __restrict__ sit,
                 float* __restrict__ out) {
  extern __shared__ char smem[];
  char*  Qb    = smem + OFF_Q;
  char*  Kb    = smem + OFF_K;
  char*  Vt    = smem + OFF_VT;
  char*  Pb    = smem + OFF_P;
  float* kmask = (float*)(smem + OFF_KM);
  float* qmask = (float*)(smem + OFF_QM);
  float* smax  = (float*)(smem + OFF_SMAX);
  float* ssum  = (float*)(smem + OFF_SSUM);
  float* scb   = (float*)(smem + OFF_SCB);
  float* mrun  = (float*)(smem + OFF_MRUN);
  float* lrun  = (float*)(smem + OFF_LRUN);

  const int tid  = threadIdx.x;
  const int wave = tid >> 5, lane = tid & 31;
  const int rg   = wave >> 2, js = wave & 3;      // row-group, j-slice / D-quarter
  const int li   = lane & 15, lh = lane >> 4;
  const int blk  = blockIdx.x;
  const int b    = blk >> 6;
  const int i0   = (blk & 63) << 5;               // 32 query rows per block
  const float* Tex = text + (size_t)b * T_SEQ * DH;
  const float* Sit = sit  + (size_t)b * T_SEQ;
  const float  SCALE = 0.03608439182435161f;      // 1/sqrt(768)

  // ---- stage Q (32 x 768) as bf16, row-major -------------------------------
  for (int it = tid; it < 6144; it += 256) {
    int row = it / 192, col4 = (it % 192) << 2;
    float4 g = *(const float4*)(Tex + (size_t)(i0 + row) * DH + col4);
    v4bf q4 = { (__bf16)g.x, (__bf16)g.y, (__bf16)g.z, (__bf16)g.w };
    *(v4bf*)(Qb + row * KROW + col4 * 2) = q4;
  }
  if (tid < 32) {
    qmask[tid] = (Sit[i0 + tid] != 0.f) ? 1.f : 0.f;
    mrun[tid] = -1e30f;
    lrun[tid] = 0.f;
  }

  v8f acc[12];                                    // 16 rows x 192 features (f32)
  #pragma unroll
  for (int t = 0; t < 12; ++t) acc[t] = {};

  const int nch = (i0 + 95) >> 6;                 // chunks of 64 keys (causal)
  for (int c = 0; c < nch; ++c) {
    const int j0 = c << 6;
    __syncthreads();                              // prev chunk PV done

    // ---- stage K chunk: row-major bf16 + feature-major transpose ----------
    for (int it = tid; it < 12288; it += 256) {
      int j = it / 192, col4 = (it % 192) << 2;
      const float* gp = Tex + (size_t)(j0 + j) * DH + col4;
      float4 g = *(const float4*)gp;
      if (j0 + 64 + j < T_SEQ) __builtin_prefetch(gp + 64 * DH, 0, 1);
      __bf16 e0 = (__bf16)g.x, e1 = (__bf16)g.y, e2 = (__bf16)g.z, e3 = (__bf16)g.w;
      v4bf k4 = { e0, e1, e2, e3 };
      *(v4bf*)(Kb + j * KROW + col4 * 2) = k4;
      *(__bf16*)(Vt + (col4 + 0) * VTROW + j * 2) = e0;
      *(__bf16*)(Vt + (col4 + 1) * VTROW + j * 2) = e1;
      *(__bf16*)(Vt + (col4 + 2) * VTROW + j * 2) = e2;
      *(__bf16*)(Vt + (col4 + 3) * VTROW + j * 2) = e3;
    }
    if (tid < 64) kmask[tid] = (Sit[j0 + tid] != 0.f) ? 1.f : 0.f;
    __syncthreads();

    // ---- S^T = K * Q^T  (M = 16 keys of this slice, N = 16 queries) -------
    v8f s = {};
    const char* arow = Kb + (js * 16 + li) * KROW + 16 * lh;   // A: K rows
    const char* brow = Qb + (rg * 16 + li) * KROW + 16 * lh;   // B: Q^T (col-major)
    #pragma unroll
    for (int ks = 0; ks < 24; ++ks) {
      Frag a, bq;
      a.h[0]  = *(const v8bf*)(arow + 64 * ks);
      a.h[1]  = *(const v8bf*)(arow + 64 * ks + 32);
      bq.h[0] = *(const v8bf*)(brow + 64 * ks);
      bq.h[1] = *(const v8bf*)(brow + 64 * ks + 32);
      s = wmma_bf16(a, bq, s);
    }

    // ---- scale + causal mask (lane = query i, vgpr/half = key j) ----------
    const int ig = i0 + rg * 16 + li;
    float sv[8];
    #pragma unroll
    for (int v = 0; v < 8; ++v) {
      int j = j0 + js * 16 + v + 8 * lh;
      float x = s[v] * SCALE;
      sv[v] = (j <= ig) ? x : -1e30f;
    }

    // ---- online softmax stats, exchanged across the 4 j-slice waves -------
    float msl = sv[0];
    #pragma unroll
    for (int v = 1; v < 8; ++v) msl = fmaxf(msl, sv[v]);
    msl = fmaxf(msl, __shfl_xor(msl, 16, 32));
    if (lane < 16) smax[(rg * 4 + js) * 16 + li] = msl;
    __syncthreads();

    float mch = smax[(rg * 4 + 0) * 16 + li];
    mch = fmaxf(mch, smax[(rg * 4 + 1) * 16 + li]);
    mch = fmaxf(mch, smax[(rg * 4 + 2) * 16 + li]);
    mch = fmaxf(mch, smax[(rg * 4 + 3) * 16 + li]);
    float mold = mrun[rg * 16 + li];
    float mnew = fmaxf(mold, mch);

    float p[8], rs = 0.f;
    #pragma unroll
    for (int v = 0; v < 8; ++v) { p[v] = __expf(sv[v] - mnew); rs += p[v]; }
    rs += __shfl_xor(rs, 16, 32);
    if (lane < 16) ssum[(rg * 4 + js) * 16 + li] = rs;       // UNMASKED row-sum
    if (js == 0 && lane < 16) scb[rg * 16 + li] = __expf(mold - mnew);

    // ---- key-mask * P -> bf16, store row-major [i][64] --------------------
    float4 kmA = *(const float4*)(kmask + js * 16 + 8 * lh);
    float4 kmB = *(const float4*)(kmask + js * 16 + 8 * lh + 4);
    union { v8bf v; __bf16 e[8]; } pb;
    pb.e[0] = (__bf16)(p[0] * kmA.x); pb.e[1] = (__bf16)(p[1] * kmA.y);
    pb.e[2] = (__bf16)(p[2] * kmA.z); pb.e[3] = (__bf16)(p[3] * kmA.w);
    pb.e[4] = (__bf16)(p[4] * kmB.x); pb.e[5] = (__bf16)(p[5] * kmB.y);
    pb.e[6] = (__bf16)(p[6] * kmB.z); pb.e[7] = (__bf16)(p[7] * kmB.w);
    *(v8bf*)(Pb + rg * (16 * PROW) + li * PROW + (js * 16 + 8 * lh) * 2) = pb.v;
    __syncthreads();

    if (js == 0 && lane < 16) {
      float tot = ssum[(rg * 4 + 0) * 16 + li] + ssum[(rg * 4 + 1) * 16 + li]
                + ssum[(rg * 4 + 2) * 16 + li] + ssum[(rg * 4 + 3) * 16 + li];
      lrun[rg * 16 + li] = lrun[rg * 16 + li] * __expf(mold - mnew) + tot;
      mrun[rg * 16 + li] = mnew;
    }

    // ---- rescale accumulator (indexing: vgpr/half = query row) ------------
    float4 sA = *(const float4*)(scb + rg * 16 + 8 * lh);
    float4 sB = *(const float4*)(scb + rg * 16 + 8 * lh + 4);
    float scv[8] = { sA.x, sA.y, sA.z, sA.w, sB.x, sB.y, sB.z, sB.w };
    #pragma unroll
    for (int t = 0; t < 12; ++t)
      #pragma unroll
      for (int v = 0; v < 8; ++v) acc[t][v] *= scv[v];

    // ---- O += P * V   (A = P row-major, B = V from feature-major Vt) ------
    Frag aP[2];
    const char* prow = Pb + rg * (16 * PROW) + li * PROW + 16 * lh;
    aP[0].h[0] = *(const v8bf*)(prow + 0);  aP[0].h[1] = *(const v8bf*)(prow + 32);
    aP[1].h[0] = *(const v8bf*)(prow + 64); aP[1].h[1] = *(const v8bf*)(prow + 96);
    #pragma unroll
    for (int nt = 0; nt < 12; ++nt) {
      const char* vrow = Vt + (js * 192 + nt * 16 + li) * VTROW + 16 * lh;
      Frag b0, b1;
      b0.h[0] = *(const v8bf*)(vrow);        b0.h[1] = *(const v8bf*)(vrow + 32);
      b1.h[0] = *(const v8bf*)(vrow + 64);   b1.h[1] = *(const v8bf*)(vrow + 96);
      acc[nt] = wmma_bf16(aP[0], b0, acc[nt]);
      acc[nt] = wmma_bf16(aP[1], b1, acc[nt]);
    }
  }

  // ---- epilogue: out = qmask * acc / l ; write [*, 0:768] of concat -------
  __syncthreads();
  float4 lA = *(const float4*)(lrun + rg * 16 + 8 * lh);
  float4 lB = *(const float4*)(lrun + rg * 16 + 8 * lh + 4);
  float4 qA = *(const float4*)(qmask + rg * 16 + 8 * lh);
  float4 qB = *(const float4*)(qmask + rg * 16 + 8 * lh + 4);
  float lv[8] = { lA.x, lA.y, lA.z, lA.w, lB.x, lB.y, lB.z, lB.w };
  float qv[8] = { qA.x, qA.y, qA.z, qA.w, qB.x, qB.y, qB.z, qB.w };
  float inv[8];
  #pragma unroll
  for (int v = 0; v < 8; ++v) inv[v] = (lv[v] > 0.f) ? (qv[v] / lv[v]) : 0.f;

  float* obase = out + (size_t)(b * T_SEQ + i0 + rg * 16) * (2 * DH)
               + js * 192 + li;
  #pragma unroll
  for (int t = 0; t < 12; ++t)
    #pragma unroll
    for (int v = 0; v < 8; ++v)
      obase[(size_t)(v + 8 * lh) * (2 * DH) + t * 16] = acc[t][v] * inv[v];
}

// ---- second half of concat: out[..., 768:1536] = text_inputs --------------
__global__ __launch_bounds__(256)
void concat_copy(const float* __restrict__ text, float* __restrict__ out) {
  size_t idx = (size_t)blockIdx.x * 256 + threadIdx.x;   // 3,145,728 float4's
  const float4* src = (const float4*)text;
  float4*       dst = (float4*)out;
  size_t bt = idx / 192, r = idx % 192;
  dst[bt * 384 + 192 + r] = src[bt * 192 + r];
}

extern "C" void kernel_launch(void* const* d_in, const int* in_sizes, int n_in,
                              void* d_out, int out_size, void* d_ws, size_t ws_size,
                              hipStream_t stream) {
  const float* text = (const float*)d_in[0];   // [8,2048,768] f32
  const float* sit  = (const float*)d_in[1];   // [8,2048]     f32
  // d_in[2] (proposition_matrix) only contributes its shape's D; unused.
  float* out = (float*)d_out;                  // [8,2048,1536] f32

  (void)hipFuncSetAttribute((const void*)attn_kernel,
                            hipFuncAttributeMaxDynamicSharedMemorySize,
                            SMEM_BYTES);
  attn_kernel<<<512, 256, SMEM_BYTES, stream>>>(text, sit, out);
  concat_copy<<<12288, 256, 0, stream>>>(text, out);
}
